// PointSetAbstractionLayer_66048007078355
// MI455X (gfx1250) — matrix operand order repaired and checked by reading, not compiled, not run on hardware
//
#include <hip/hip_runtime.h>

// ---------------------------------------------------------------------------
// PointNet++ Set Abstraction for MI455X (gfx1250, wave32, WMMA)
//   b=16, n=4096, c=64, ns=1024, knn=32
//   out = (centroids (16,1024,3), features (16,1024,128)) concat-flat, f32
// Strategy: fold BN into f16 weights, run all three 1x1-conv layers as
// wave-level GEMMs with v_wmma_f32_16x16x32_f16, one group (32 neighbors)
// per wave, activations staged in per-wave LDS tiles.
// Round 2: cache A-fragments across M-tiles (each weight fragment loaded from
// LDS exactly once per group), scalarize wave-uniform loads via readfirstlane,
// vectorize epilogues for packed VALU.
// ---------------------------------------------------------------------------

#define BN_EPS    1e-5f
#define INV_RAD   5.0f      // 1 / 0.2

typedef _Float16 v16h __attribute__((ext_vector_type(16)));
typedef float    v8f  __attribute__((ext_vector_type(8)));
typedef float    v4f  __attribute__((ext_vector_type(4)));
typedef float    v2f  __attribute__((ext_vector_type(2)));
typedef _Float16 h8   __attribute__((ext_vector_type(8)));
typedef _Float16 h4   __attribute__((ext_vector_type(4)));

union F16Frag { v16h v; h8 h[2]; };

constexpr int BATCH = 16;
constexpr int NPTS  = 4096;
constexpr int CIN   = 64;
constexpr int NSAMP = 1024;
constexpr int KNN   = 32;
constexpr int NG    = BATCH * NSAMP;     // 16384 groups
constexpr int CEN   = NG * 3;            // floats of centroid output
constexpr int NWAVE = 8;                 // waves per block
constexpr int KP1   = 96;                // layer-1 padded K (and x-row stride)

// A fragment: 16x32 f16, per ISA 16-bit A-matrix table.
// lanes 0-15: row m, K = [kk..kk+8) and [kk+16..kk+24)
// lanes 16-31: row m, K = [kk+8..kk+16) and [kk+24..kk+32)
__device__ __forceinline__ v16h load_afrag(const _Float16* base, int stride,
                                           int mt, int kk, int lane) {
  const int m  = mt * 16 + (lane & 15);
  const int k0 = kk + ((lane & 16) ? 8 : 0);
  F16Frag f;
  f.h[0] = *(const h8*)(base + m * stride + k0);
  f.h[1] = *(const h8*)(base + m * stride + k0 + 16);
  return f.v;
}

// B fragment: 32x16 f16 (K x N); weights stored row-major (o, Kpad) == N-major,
// so each lane's 16 K-values are two contiguous runs of 8.
// lanes 0-15: col n, K = [kk..kk+8) and [kk+16..kk+24); lanes 16-31: col n, K = [kk+8..kk+16) and [kk+24..kk+32)
__device__ __forceinline__ v16h load_bfrag(const _Float16* w, int stride,
                                           int nt, int kk, int lane) {
  const int o  = nt * 16 + (lane & 15);
  const int kb = kk + ((lane & 16) ? 8 : 0);
  F16Frag f;
  f.h[0] = *(const h8*)(w + o * stride + kb);
  f.h[1] = *(const h8*)(w + o * stride + kb + 16);
  return f.v;
}

__device__ __forceinline__ v8f wmma16(v16h a, v16h b, v8f c) {
  return __builtin_amdgcn_wmma_f32_16x16x32_f16(false, a, false, b,
                                                (short)0, c, false, false);
}

// D tile (16x16 f32): VGPR r holds (M = mt*16 + r + (lane<16?0:8), N = nt*16 + lane%16).
// Apply fused BN bias + ReLU (packed VALU), write f16 row-major activation.
__device__ __forceinline__ void store_htile(_Float16* hb, int stride, int mt,
                                            int nt, int lane, v8f acc,
                                            const float* beta) {
  const int n  = nt * 16 + (lane & 15);
  const int mb = mt * 16 + ((lane & 16) ? 8 : 0);
  const v8f zero = {};
  v8f v = acc + beta[n];                        // splat add -> v_pk_add_f32
  v = __builtin_elementwise_max(v, zero);
  _Float16* p = hb + mb * stride + n;
#pragma unroll
  for (int r = 0; r < 8; ++r) p[r * stride] = (_Float16)v[r];
}

extern "C" __global__ __launch_bounds__(256, 1) void psa_wmma_kernel(
    const float* __restrict__ points, const float* __restrict__ features,
    const int* __restrict__ fps, const int* __restrict__ nbr,
    const float* __restrict__ W1, const float* __restrict__ b1,
    const float* __restrict__ g1, const float* __restrict__ bt1,
    const float* __restrict__ mu1, const float* __restrict__ var1,
    const float* __restrict__ W2, const float* __restrict__ b2,
    const float* __restrict__ g2, const float* __restrict__ bt2,
    const float* __restrict__ mu2, const float* __restrict__ var2,
    const float* __restrict__ W3, const float* __restrict__ b3,
    const float* __restrict__ g3, const float* __restrict__ bt3,
    const float* __restrict__ mu3, const float* __restrict__ var3,
    float* __restrict__ out) {
  __shared__ _Float16 s_w1[64 * KP1];     // layer1 weights, BN-folded, cols reordered
  __shared__ _Float16 s_w2[64 * 64];
  __shared__ _Float16 s_w3[128 * 64];
  __shared__ float    sbeta1[64], sbeta2[64], sbeta3[128];
  __shared__ _Float16 s_x[NWAVE * 32 * KP1];   // per-wave input / layer-2 output
  __shared__ _Float16 s_h[NWAVE * 32 * 64];    // per-wave layer-1 output

  const int tid = threadIdx.x;

  // ---- fold BatchNorm into f16 weights + f32 bias (per block) ----
  // layer1 column reorder: x cols [0..63]=features, [64..66]=xyz, [67..95]=0
  for (int idx = tid; idx < 64 * KP1; idx += blockDim.x) {
    const int o = idx / KP1;
    const int k = idx - o * KP1;
    const float s = g1[o] * rsqrtf(var1[o] + BN_EPS);
    float w = 0.f;
    if (k < 64)      w = W1[o * 67 + 3 + k];
    else if (k < 67) w = W1[o * 67 + (k - 64)];
    s_w1[idx] = (_Float16)(w * s);
  }
  for (int idx = tid; idx < 64 * 64; idx += blockDim.x) {
    const int o = idx >> 6;
    const float s = g2[o] * rsqrtf(var2[o] + BN_EPS);
    s_w2[idx] = (_Float16)(W2[idx] * s);
  }
  for (int idx = tid; idx < 128 * 64; idx += blockDim.x) {
    const int o = idx >> 6;
    const float s = g3[o] * rsqrtf(var3[o] + BN_EPS);
    s_w3[idx] = (_Float16)(W3[idx] * s);
  }
  for (int o = tid; o < 64; o += blockDim.x) {
    const float s1v = g1[o] * rsqrtf(var1[o] + BN_EPS);
    sbeta1[o] = (b1[o] - mu1[o]) * s1v + bt1[o];
    const float s2v = g2[o] * rsqrtf(var2[o] + BN_EPS);
    sbeta2[o] = (b2[o] - mu2[o]) * s2v + bt2[o];
  }
  for (int o = tid; o < 128; o += blockDim.x) {
    const float s3v = g3[o] * rsqrtf(var3[o] + BN_EPS);
    sbeta3[o] = (b3[o] - mu3[o]) * s3v + bt3[o];
  }
  __syncthreads();

  const int wave = tid >> 5;
  const int lane = tid & 31;
  _Float16* sx = s_x + wave * 32 * KP1;
  _Float16* sh = s_h + wave * 32 * 64;

  for (int g0 = blockIdx.x * NWAVE + wave; g0 < NG; g0 += gridDim.x * NWAVE) {
    // g is wave-uniform by construction; make the compiler see it so that
    // fps/centroid reads become SMEM loads.
    const int g  = __builtin_amdgcn_readfirstlane(g0);
    const int bb = g >> 10;                       // / NSAMP
    const float* pbase = points + (size_t)bb * NPTS * 3;

    // centroid (uniform -> s_load)
    const int fidx = fps[g];
    const float cx = pbase[fidx * 3 + 0];
    const float cy = pbase[fidx * 3 + 1];
    const float cz = pbase[fidx * 3 + 2];
    if (lane < 3) out[(size_t)g * 3 + lane] = (lane == 0) ? cx : (lane == 1) ? cy : cz;

    // ---- gather: each lane builds one neighbor row of X (32 x 96 f16) ----
    const int ni = nbr[(size_t)g * KNN + lane];
    const float* frow = features + ((size_t)bb * NPTS + ni) * CIN;
    _Float16* xrow = sx + lane * KP1;
#pragma unroll
    for (int i = 0; i < 16; ++i) {               // 16 x global_load_b128
      const float4 f = ((const float4*)frow)[i];
      h4 hv = { (_Float16)f.x, (_Float16)f.y, (_Float16)f.z, (_Float16)f.w };
      *(h4*)(xrow + 4 * i) = hv;                 // ds_store_b64, 8B aligned
    }
    const float px = pbase[ni * 3 + 0];
    const float py = pbase[ni * 3 + 1];
    const float pz = pbase[ni * 3 + 2];
    h4 gp = { (_Float16)((px - cx) * INV_RAD), (_Float16)((py - cy) * INV_RAD),
              (_Float16)((pz - cz) * INV_RAD), (_Float16)0.f };
    *(h4*)(xrow + 64) = gp;
    h4 zz = { (_Float16)0.f, (_Float16)0.f, (_Float16)0.f, (_Float16)0.f };
#pragma unroll
    for (int i = 0; i < 7; ++i) *(h4*)(xrow + 68 + 4 * i) = zz;  // pad K 68..95

    // ---- Layer 1: X(32x96) @ W1'(64x96)^T -> H1(32x64) ---- 24 WMMAs
    {
      v16h a[2][3];
#pragma unroll
      for (int mt = 0; mt < 2; ++mt)
#pragma unroll
        for (int kc = 0; kc < 3; ++kc)
          a[mt][kc] = load_afrag(sx, KP1, mt, kc * 32, lane);
#pragma unroll
      for (int nt = 0; nt < 4; ++nt) {
        const v16h bw0 = load_bfrag(s_w1, KP1, nt, 0, lane);
        const v16h bw1 = load_bfrag(s_w1, KP1, nt, 32, lane);
        const v16h bw2 = load_bfrag(s_w1, KP1, nt, 64, lane);
#pragma unroll
        for (int mt = 0; mt < 2; ++mt) {
          v8f acc = {};
          acc = wmma16(a[mt][0], bw0, acc);
          acc = wmma16(a[mt][1], bw1, acc);
          acc = wmma16(a[mt][2], bw2, acc);
          store_htile(sh, 64, mt, nt, lane, acc, sbeta1);
        }
      }
    }

    // ---- Layer 2: H1(32x64) @ W2'(64x64)^T -> H2 (into sx) ---- 16 WMMAs
    {
      v16h a[2][2];
#pragma unroll
      for (int mt = 0; mt < 2; ++mt)
#pragma unroll
        for (int kc = 0; kc < 2; ++kc)
          a[mt][kc] = load_afrag(sh, 64, mt, kc * 32, lane);
#pragma unroll
      for (int nt = 0; nt < 4; ++nt) {
        const v16h bw0 = load_bfrag(s_w2, 64, nt, 0, lane);
        const v16h bw1 = load_bfrag(s_w2, 64, nt, 32, lane);
#pragma unroll
        for (int mt = 0; mt < 2; ++mt) {
          v8f acc = {};
          acc = wmma16(a[mt][0], bw0, acc);
          acc = wmma16(a[mt][1], bw1, acc);
          store_htile(sx, KP1, mt, nt, lane, acc, sbeta2);  // cols 0..63 of sx
        }
      }
    }

    // ---- Layer 3: H2(32x64) @ W3'(128x64)^T, fused max over 32 rows ----
    const v16h a00 = load_afrag(sx, KP1, 0, 0, lane);
    const v16h a01 = load_afrag(sx, KP1, 0, 32, lane);
    const v16h a10 = load_afrag(sx, KP1, 1, 0, lane);
    const v16h a11 = load_afrag(sx, KP1, 1, 32, lane);
    float* orow = out + CEN + (size_t)g * 128;
#pragma unroll
    for (int nt = 0; nt < 8; ++nt) {             // 32 WMMAs
      const v16h bw0 = load_bfrag(s_w3, 64, nt, 0, lane);
      const v16h bw1 = load_bfrag(s_w3, 64, nt, 32, lane);
      v8f c0 = {}, c1 = {};
      c0 = wmma16(a00, bw0, c0);
      c0 = wmma16(a01, bw1, c0);
      c1 = wmma16(a10, bw0, c1);
      c1 = wmma16(a11, bw1, c1);
      // max over the 16 row-values this lane holds (packed VALU tree)
      const v8f m8 = __builtin_elementwise_max(c0, c1);
      const v4f m4 = __builtin_elementwise_max(
          __builtin_shufflevector(m8, m8, 0, 1, 2, 3),
          __builtin_shufflevector(m8, m8, 4, 5, 6, 7));
      const v2f m2 = __builtin_elementwise_max(
          __builtin_shufflevector(m4, m4, 0, 1),
          __builtin_shufflevector(m4, m4, 2, 3));
      float m = fmaxf(m2[0], m2[1]);
      // combine lanes L and L+16 (same output channel, other row half)
      m = fmaxf(m, __shfl_xor(m, 16, 32));
      const int n = nt * 16 + (lane & 15);
      float v = m + sbeta3[n];                   // max commutes with +beta, relu
      v = v > 0.f ? v : 0.f;
      if (lane < 16) orow[n] = v;
    }
  }
}

extern "C" void kernel_launch(void* const* d_in, const int* in_sizes, int n_in,
                              void* d_out, int out_size, void* d_ws, size_t ws_size,
                              hipStream_t stream) {
  (void)in_sizes; (void)n_in; (void)out_size; (void)d_ws; (void)ws_size;
  const float* points   = (const float*)d_in[0];
  const float* features = (const float*)d_in[1];
  const int*   fps      = (const int*)d_in[2];
  const int*   nbr      = (const int*)d_in[3];
  const float* W1 = (const float*)d_in[4];
  const float* b1 = (const float*)d_in[5];
  const float* g1 = (const float*)d_in[6];
  const float* bt1 = (const float*)d_in[7];
  const float* mu1 = (const float*)d_in[8];
  const float* var1 = (const float*)d_in[9];
  const float* W2 = (const float*)d_in[10];
  const float* b2 = (const float*)d_in[11];
  const float* g2 = (const float*)d_in[12];
  const float* bt2 = (const float*)d_in[13];
  const float* mu2 = (const float*)d_in[14];
  const float* var2 = (const float*)d_in[15];
  const float* W3 = (const float*)d_in[16];
  const float* b3 = (const float*)d_in[17];
  const float* g3 = (const float*)d_in[18];
  const float* bt3 = (const float*)d_in[19];
  const float* mu3 = (const float*)d_in[20];
  const float* var3 = (const float*)d_in[21];
  float* out = (float*)d_out;

  dim3 grid(1024);   // 1024 blocks * 8 waves = 8192 wave-slots -> 2 groups/wave
  dim3 block(256);   // 8 wave32 waves
  hipLaunchKernelGGL(psa_wmma_kernel, grid, block, 0, stream,
                     points, features, fps, nbr,
                     W1, b1, g1, bt1, mu1, var1,
                     W2, b2, g2, bt2, mu2, var2,
                     W3, b3, g3, bt3, mu3, var3,
                     out);
}
